// BilinearLayer_60859686585068
// MI455X (gfx1250) — compile-verified, matrix operand order checked by reading, not run on hardware
//
#include <hip/hip_runtime.h>
#include <hip/hip_bf16.h>

typedef __bf16 v16bf __attribute__((ext_vector_type(16)));
typedef __bf16 v8bf  __attribute__((ext_vector_type(8)));
typedef float  v8f   __attribute__((ext_vector_type(8)));
typedef float  v4f   __attribute__((ext_vector_type(4)));

#define BR_ 64     // B*R
#define N_  1024
#define F_  256

// Build a 16x32 bf16 WMMA fragment (A- or B-style: lane<16 -> K 0..7,16..23;
// lane>=16 -> K 8..15,24..31 -- caller pre-offsets pointer by 8 elems for hi lanes)
// from fp32 memory, converting to bf16 in registers.
__device__ __forceinline__ v16bf frag_from_f32(const float* __restrict__ p) {
  v4f x0 = *(const v4f*)(p);
  v4f x1 = *(const v4f*)(p + 4);
  v4f x2 = *(const v4f*)(p + 16);
  v4f x3 = *(const v4f*)(p + 20);
  v16bf a;
#pragma unroll
  for (int i = 0; i < 4; ++i) {
    a[i]      = (__bf16)x0[i];
    a[4 + i]  = (__bf16)x1[i];
    a[8 + i]  = (__bf16)x2[i];
    a[12 + i] = (__bf16)x3[i];
  }
  return a;
}

// Same fragment shape from contiguous bf16 memory (global ws or LDS).
__device__ __forceinline__ v16bf frag_from_bf16(const __bf16* __restrict__ p) {
  v8bf x0 = *(const v8bf*)(p);
  v8bf x1 = *(const v8bf*)(p + 16);
  v16bf a;
#pragma unroll
  for (int i = 0; i < 8; ++i) { a[i] = x0[i]; a[8 + i] = x1[i]; }
  return a;
}

// ---------------- Phase 1: T16[br][n][g] = bf16( E[br] @ Bs[r] ) ----------------
// grid: (64, 8) ; block: 256 threads (8 waves, each wave -> 16 rows of T)
__global__ __launch_bounds__(256) void bilinear_phase1(
    const float* __restrict__ inputs,   // [BR][N][F] fp32
    const float* __restrict__ Bs,       // [R][F][F]  fp32
    __bf16* __restrict__ T16)           // [BR][N][F] bf16 (workspace)
{
  extern __shared__ __bf16 Bt[];        // [F][F] = Bs[r] transposed: Bt[g*F + f]
  const int br    = blockIdx.x;
  const int r     = br & 7;
  const int ntile = blockIdx.y;         // 128-row tile
  const int tid   = threadIdx.x;
  const int lane  = tid & 31;
  const int wave  = tid >> 5;
  const int l15   = lane & 15;
  const int sel   = (lane >> 4) ? 8 : 0;   // hi half-wave reads K+8

  // Stage Bs[r] transposed into LDS as bf16: coalesced b128 global reads.
  const float* Brsrc = Bs + (size_t)r * F_ * F_;
#pragma unroll 4
  for (int i = 0; i < 64; ++i) {
    int idx = i * 1024 + tid * 4;       // flat into [F][F]
    int f = idx >> 8;
    int g = idx & 255;
    v4f x = *(const v4f*)(Brsrc + idx);
#pragma unroll
    for (int j = 0; j < 4; ++j) Bt[(g + j) * F_ + f] = (__bf16)x[j];
  }
  __syncthreads();

  // Preload this wave's 16 rows of E as 8 bf16 K-fragments (K=256).
  const int rowblk = ntile * 128 + wave * 16;
  const float* Arow = inputs + ((size_t)br * N_ + rowblk + l15) * F_ + sel;
  v16bf A[8];
#pragma unroll
  for (int kc = 0; kc < 8; ++kc) A[kc] = frag_from_f32(Arow + kc * 32);

  __bf16* Trow = T16 + ((size_t)br * N_ + rowblk) * F_;
  const int mbase = (lane >> 4) ? 8 : 0;

  for (int gc = 0; gc < 16; ++gc) {     // 16 column tiles of 16
    v8f acc = {};
    const __bf16* Bp = Bt + (size_t)(gc * 16 + l15) * F_ + sel;
#pragma unroll
    for (int kc = 0; kc < 8; ++kc) {
      v16bf b = frag_from_bf16(Bp + kc * 32);   // ds_load_b128 x2
      acc = __builtin_amdgcn_wmma_f32_16x16x32_bf16(
          false, A[kc], false, b, (short)0, acc, false, false);
    }
    const int g = gc * 16 + l15;
#pragma unroll
    for (int j = 0; j < 8; ++j)
      Trow[(size_t)(mbase + j) * F_ + g] = (__bf16)acc[j];
  }
}

// ---------------- Phase 2: S = sigmoid( T16 @ E^T ) ----------------
// grid: (64, 8, 8) ; block: 256 threads. Workgroup -> 128x128 output tile,
// wave -> 64x32 tile (4x2 fragments).
__global__ __launch_bounds__(256) void bilinear_phase2(
    const float* __restrict__ inputs,   // [BR][N][F] fp32 (E, reused as B=E^T)
    const __bf16* __restrict__ T16,     // [BR][N][F] bf16
    float* __restrict__ out)            // [BR][N][N] fp32
{
  const int br   = blockIdx.x;
  const int nt   = blockIdx.y;          // 128-row tile (n)
  const int mt   = blockIdx.z;          // 128-col tile (m)
  const int tid  = threadIdx.x;
  const int lane = tid & 31;
  const int wave = tid >> 5;
  const int wrow = wave & 1;            // 2 x 64 rows
  const int wcol = wave >> 1;           // 4 x 32 cols
  const int l15  = lane & 15;
  const int sel  = (lane >> 4) ? 8 : 0;

  const size_t base = (size_t)br * N_;
  const __bf16* Abase = T16 + (base + nt * 128 + wrow * 64 + l15) * F_ + sel;
  const float*  Bbase = inputs + (base + mt * 128 + wcol * 32 + l15) * F_ + sel;

  v8f acc[4][2] = {};

  for (int kc = 0; kc < 8; ++kc) {      // K = 256 in steps of 32
    v16bf a[4], b[2];
#pragma unroll
    for (int i = 0; i < 4; ++i)
      a[i] = frag_from_bf16(Abase + (size_t)i * 16 * F_ + kc * 32);
#pragma unroll
    for (int j = 0; j < 2; ++j)
      b[j] = frag_from_f32(Bbase + (size_t)j * 16 * F_ + kc * 32);
    if (kc < 7) {                       // prefetch next K-chunk of A
      __builtin_prefetch(Abase + (kc + 1) * 32, 0, 1);
    }
#pragma unroll
    for (int i = 0; i < 4; ++i)
#pragma unroll
      for (int j = 0; j < 2; ++j)
        acc[i][j] = __builtin_amdgcn_wmma_f32_16x16x32_bf16(
            false, a[i], false, b[j], (short)0, acc[i][j], false, false);
  }

  // sigmoid + coalesced fp32 stores (lane = output column m)
  const int mbase = (lane >> 4) ? 8 : 0;
  float* Obase = out + (base + nt * 128 + wrow * 64) * (size_t)N_
                     + mt * 128 + wcol * 32;
#pragma unroll
  for (int i = 0; i < 4; ++i)
#pragma unroll
    for (int j = 0; j < 2; ++j)
#pragma unroll
      for (int v = 0; v < 8; ++v) {
        float x = acc[i][j][v];
        float s = 1.0f / (1.0f + __expf(-x));
        Obase[(size_t)(i * 16 + mbase + v) * N_ + j * 16 + l15] = s;
      }
}

extern "C" void kernel_launch(void* const* d_in, const int* in_sizes, int n_in,
                              void* d_out, int out_size, void* d_ws, size_t ws_size,
                              hipStream_t stream) {
  const float* inputs = (const float*)d_in[0];   // [8,8,1024,256] fp32
  const float* Bs     = (const float*)d_in[1];   // [8,256,256]    fp32
  float* out = (float*)d_out;                    // [8,8,1024,1024] fp32
  __bf16* T16 = (__bf16*)d_ws;                   // 64*1024*256*2 = 33.5 MB

  dim3 g1(BR_, N_ / 128);
  bilinear_phase1<<<g1, 256, F_ * F_ * sizeof(__bf16), stream>>>(inputs, Bs, T16);

  dim3 g2(BR_, N_ / 128, N_ / 128);
  bilinear_phase2<<<g2, 256, 0, stream>>>(inputs, T16, out);
}
